// SDEnet_10625749090817
// MI455X (gfx1250) — matrix operand order repaired and checked by reading, not compiled, not run on hardware
//
#include <hip/hip_runtime.h>

// Problem constants (match the reference)
#define B_ROWS   32768
#define D_IN     100
#define D_DIM    1024
#define NBLK     5
#define H_CONST  0.1f
#define SQRT_H   0.31622776601683794f
#define EPS_CONST 1e-4f

// Stage B tiles with the Tensor Data Mover (CDNA5 TDM). Set to 0 to fall back
// to manual vectorized global->LDS staging.
#define USE_TDM_B 1

typedef __bf16 v8bf  __attribute__((ext_vector_type(8)));
typedef __bf16 v16bf __attribute__((ext_vector_type(16)));
typedef float  v8f   __attribute__((ext_vector_type(8)));

typedef unsigned int u32x4 __attribute__((ext_vector_type(4)));
typedef int          i32x4 __attribute__((ext_vector_type(4)));
typedef int          i32x8 __attribute__((ext_vector_type(8)));

// round-to-nearest-even f32 -> bf16 bit pattern
__device__ __forceinline__ unsigned short f2bf(float f) {
  unsigned int u = __float_as_uint(f);
  unsigned int r = (u + 0x7FFFu + ((u >> 16) & 1u)) >> 16;
  return (unsigned short)r;
}

// Build a 16-element bf16 fragment from two 16-byte LDS chunks
__device__ __forceinline__ v16bf frag16(const unsigned short* p0, const unsigned short* p1) {
  v8bf lo = *(const v8bf*)p0;
  v8bf hi = *(const v8bf*)p1;
  return __builtin_shufflevector(lo, hi, 0,1,2,3,4,5,6,7,8,9,10,11,12,13,14,15);
}

#if USE_TDM_B
// TDM: DMA a 32(k) x 128(n) bf16 tile (row stride 1024 elems) into LDS.
// Descriptor bit layout per CDNA5 ISA ch.8 (D# groups 0/1; groups 2/3 unused for 2D).
__device__ __forceinline__ void tdm_load_b_tile(const unsigned short* gsrc,
                                                unsigned lds_off) {
  unsigned long long ga = (unsigned long long)gsrc;
  u32x4 g0;
  g0[0] = 1u;                                              // count=1, user D#
  g0[1] = lds_off;                                         // lds_addr (bytes)
  g0[2] = (unsigned)(ga & 0xFFFFFFFFu);                    // global_addr[31:0]
  g0[3] = (unsigned)((ga >> 32) & 0x01FFFFFFu) | (2u << 30); // addr[56:32] | type=2
  i32x8 g1;
  g1[0] = (int)(1u << 16);                 // workgroup_mask=0, data_size=1 (2B)
  g1[1] = (int)((1024u & 0xFFFFu) << 16);  // tensor_dim0[15:0] -> bits[63:48]
  g1[2] = (int)(((1024u >> 16) & 0xFFFFu) | ((1024u & 0xFFFFu) << 16)); // dim0 hi | dim1 lo
  g1[3] = (int)(((1024u >> 16) & 0xFFFFu) | (128u << 16)); // dim1 hi | tile_dim0=128
  g1[4] = (int)32u;                        // tile_dim1=32, tile_dim2=0
  g1[5] = (int)1024u;                      // tensor_dim0_stride[31:0] = 1024 elems
  g1[6] = 0;                               // dim0_stride hi | dim1_stride lo
  g1[7] = 0;                               // dim1_stride hi
  i32x4 gz = {0, 0, 0, 0};
#if defined(__clang_major__) && (__clang_major__ >= 23)
  i32x8 gz8 = {0, 0, 0, 0, 0, 0, 0, 0};
  __builtin_amdgcn_tensor_load_to_lds(g0, g1, gz, gz, gz8, 0);
#else
  __builtin_amdgcn_tensor_load_to_lds(g0, g1, gz, gz, 0);
#endif
}
#endif

// ---------------- conversion kernels (run once per launch) ----------------

// x [B,100] fp32 -> x_bf [B,128] bf16, zero-padded K
__global__ __launch_bounds__(256) void k_conv_x(const float* __restrict__ x,
                                                unsigned short* __restrict__ xb) {
  const int total = B_ROWS * 128;
  for (int i = blockIdx.x * blockDim.x + threadIdx.x; i < total;
       i += gridDim.x * blockDim.x) {
    int b = i >> 7, k = i & 127;
    float v = (k < D_IN) ? x[b * D_IN + k] : 0.0f;
    xb[i] = f2bf(v);
  }
}

// W0 [1024,100] fp32 -> w0t [128,1024] bf16 K-major (zero-padded K)
__global__ __launch_bounds__(256) void k_conv_w0(const float* __restrict__ W0,
                                                 unsigned short* __restrict__ w0t) {
  const int total = 128 * D_DIM;
  for (int i = blockIdx.x * blockDim.x + threadIdx.x; i < total;
       i += gridDim.x * blockDim.x) {
    int k = i >> 10, n = i & 1023;
    float v = (k < D_IN) ? W0[n * D_IN + k] : 0.0f;
    w0t[i] = f2bf(v);
  }
}

// Ws [5,1024,1024] fp32 row-major [n][k] -> wst [5,1024,1024] bf16 K-major [k][n]
__global__ __launch_bounds__(256) void k_conv_ws(const float* __restrict__ Ws,
                                                 unsigned short* __restrict__ wst) {
  const size_t total = (size_t)NBLK * D_DIM * D_DIM;
  for (size_t i = (size_t)blockIdx.x * blockDim.x + threadIdx.x; i < total;
       i += (size_t)gridDim.x * blockDim.x) {
    size_t blk = i / ((size_t)D_DIM * D_DIM);
    size_t rem = i - blk * (size_t)D_DIM * D_DIM;
    int k = (int)(rem >> 10);
    int n = (int)(rem & 1023);
    wst[i] = f2bf(Ws[blk * (size_t)D_DIM * D_DIM + (size_t)n * D_DIM + k]);
  }
}

// ---------------- per-row dot products S[i][b] = noise[i][b] . bw[i] ----------------
__global__ __launch_bounds__(256) void k_dots(const float* __restrict__ noise,
                                              const float* __restrict__ bw,
                                              float* __restrict__ S) {
  int wid = threadIdx.x >> 5, lane = threadIdx.x & 31;
  int row = blockIdx.x * 8 + wid;            // row in [0, 5*32768)
  if (row >= NBLK * B_ROWS) return;
  const float* zp = noise + (size_t)row * D_DIM;
  const float* wp = bw + (row >> 15) * D_DIM;   // row>>15 == block index
  float s = 0.0f;
#pragma unroll
  for (int c = lane * 4; c < D_DIM; c += 128) {
    float4 zv = *(const float4*)(zp + c);
    float4 wv = *(const float4*)(wp + c);
    s += zv.x * wv.x + zv.y * wv.y + zv.z * wv.z + zv.w * wv.w;
  }
#pragma unroll
  for (int off = 16; off > 0; off >>= 1) s += __shfl_down(s, off, 32);
  if (lane == 0) S[row] = s;
}

// ---------------- final GEMV: y[b] = out[b] . Wlast + blast ----------------
__global__ __launch_bounds__(256) void k_gemv(const float* __restrict__ out,
                                              const float* __restrict__ Wlast,
                                              const float* __restrict__ blast,
                                              float* __restrict__ y) {
  int wid = threadIdx.x >> 5, lane = threadIdx.x & 31;
  int row = blockIdx.x * 8 + wid;
  if (row >= B_ROWS) return;
  const float* op = out + (size_t)row * D_DIM;
  float s = 0.0f;
#pragma unroll
  for (int c = lane * 4; c < D_DIM; c += 128) {
    float4 a = *(const float4*)(op + c);
    float4 w = *(const float4*)(Wlast + c);
    s += a.x * w.x + a.y * w.y + a.z * w.z + a.w * w.w;
  }
#pragma unroll
  for (int off = 16; off > 0; off >>= 1) s += __shfl_down(s, off, 32);
  if (lane == 0) y[row] = s + blast[0];
}

// ---------------- WMMA GEMM, 128x128 tile, BK=32, 8 waves ----------------
// A_BF16:   A is bf16 [M][lda]  (gemm0: x_bf, lda=128, K=128)
//           else A is fp32 [M][lda] (block GEMM: out, lda=1024, K=1024)
// BLOCK_EPI: out = prev + H*tanh(acc+bias) + sqrt(H)*(S[m]*bw[n] + eps*noise)
//            else out = tanh(acc + bias)
template <bool A_BF16, bool BLOCK_EPI>
__global__ __launch_bounds__(256) void k_gemm(const void* __restrict__ Aptr,
                                              const unsigned short* __restrict__ Bmat,
                                              const float* __restrict__ bias,
                                              const float* __restrict__ Svec,
                                              const float* __restrict__ bwv,
                                              const float* __restrict__ noise,
                                              float* __restrict__ Out,
                                              int K, int lda) {
  __shared__ unsigned short lA[128 * 32];   // [row][k]  8 KB
  __shared__ unsigned short lB[32 * 128];   // [k][n]    8 KB

  const int tid = threadIdx.x;
  const int blockN = blockIdx.x * 128;
  const int blockM = blockIdx.y * 128;
  const int lane = tid & 31;
  const int wid = tid >> 5;
  const int wm = wid & 3;          // 4 waves along M (32 rows each)
  const int wn = wid >> 2;         // 2 waves along N (64 cols each)
  const int half = lane >> 4;      // lane group
  const int ln = lane & 15;
  const int abase = half * 8;      // A-fragment K base per lane group

#if USE_TDM_B
  const unsigned ldsB_off = (unsigned)((const char*)lB - (const char*)lA);
#endif

  v8f zero8 = {0.f, 0.f, 0.f, 0.f, 0.f, 0.f, 0.f, 0.f};
  v8f acc[2][4];
#pragma unroll
  for (int i = 0; i < 2; ++i)
#pragma unroll
    for (int j = 0; j < 4; ++j) acc[i][j] = zero8;

  for (int k0 = 0; k0 < K; k0 += 32) {
    __syncthreads();
    // -------- stage B tile (32x128 bf16, K-major in global) --------
#if USE_TDM_B
    if (wid == 0) {
      tdm_load_b_tile(Bmat + (size_t)k0 * D_DIM + blockN, ldsB_off);
    }
#else
    {
      int kk = tid >> 4;
      int col8 = (tid & 15) << 3;
#pragma unroll
      for (int p = 0; p < 2; ++p) {
        int k = kk + p * 16;
        const unsigned short* src = Bmat + (size_t)(k0 + k) * D_DIM + blockN + col8;
        *(uint4*)(lB + k * 128 + col8) = *(const uint4*)src;
      }
    }
#endif
    // -------- stage A tile (128x32 bf16) --------
    if (A_BF16) {
      const unsigned short* A = (const unsigned short*)Aptr;
      int row = tid >> 2;
      int col8 = (tid & 3) << 3;
#pragma unroll
      for (int p = 0; p < 2; ++p) {
        int r = row + p * 64;
        const unsigned short* src = A + (size_t)(blockM + r) * lda + k0 + col8;
        *(uint4*)(lA + r * 32 + col8) = *(const uint4*)src;
      }
    } else {
      const float* A = (const float*)Aptr;
      int row = tid >> 3;
      int col4 = (tid & 7) << 2;
#pragma unroll
      for (int p = 0; p < 4; ++p) {
        int r = row + p * 32;
        const float* src = A + (size_t)(blockM + r) * lda + k0 + col4;
        float4 v = *(const float4*)src;
        ushort4 h;
        h.x = f2bf(v.x); h.y = f2bf(v.y); h.z = f2bf(v.z); h.w = f2bf(v.w);
        *(ushort4*)(lA + r * 32 + col4) = h;
      }
      if (k0 + 32 < K)  // gfx1250 global_prefetch_b8 for the next A tile
        __builtin_prefetch((const void*)(A + (size_t)(blockM + row) * lda + k0 + 32 + col4), 0, 0);
    }
#if USE_TDM_B
    if (wid == 0) {
      __builtin_amdgcn_s_wait_tensorcnt(0);   // drain TDM before barrier release
    }
#endif
    __syncthreads();

    // -------- fragments + 2x4 WMMA --------
    v16bf afr[2];
#pragma unroll
    for (int tmi = 0; tmi < 2; ++tmi) {
      int row = wm * 32 + tmi * 16 + ln;    // A: M = lane%16
      afr[tmi] = frag16(lA + row * 32 + abase, lA + row * 32 + abase + 16);
    }
    v16bf bfr[4];
#pragma unroll
    for (int tni = 0; tni < 4; ++tni) {
      int n0 = wn * 64 + tni * 16;          // B: N = lane%16, K in table order (same as A)
#pragma unroll
      for (int s = 0; s < 16; ++s) {
        int kb = (s < 8) ? (abase + s) : (16 + abase + (s - 8));
        bfr[tni][s] = ((const __bf16*)lB)[kb * 128 + n0 + ln];
      }
    }
#pragma unroll
    for (int tmi = 0; tmi < 2; ++tmi)
#pragma unroll
      for (int tni = 0; tni < 4; ++tni)
        acc[tmi][tni] = __builtin_amdgcn_wmma_f32_16x16x32_bf16(
            false, afr[tmi], false, bfr[tni], (short)0, acc[tmi][tni], false, false);
  }

  // -------- fused epilogue --------
#pragma unroll
  for (int tmi = 0; tmi < 2; ++tmi) {
#pragma unroll
    for (int tni = 0; tni < 4; ++tni) {
      int mbase = blockM + wm * 32 + tmi * 16 + half * 8;  // C: VGPR r -> M=r(+8)
      int n = blockN + wn * 64 + tni * 16 + ln;            // C: N = lane%16
#pragma unroll
      for (int r = 0; r < 8; ++r) {
        int m = mbase + r;
        size_t idx = (size_t)m * D_DIM + n;
        float v = acc[tmi][tni][r];
        if (BLOCK_EPI) {
          const float* prev = (const float*)Aptr;  // fp32 'out' is also A
          Out[idx] = prev[idx] + H_CONST * tanhf(v + bias[n]) +
                     SQRT_H * (Svec[m] * bwv[n] + EPS_CONST * noise[idx]);
        } else {
          Out[idx] = tanhf(v + bias[n]);
        }
      }
    }
  }
}

// ---------------- host launch ----------------
extern "C" void kernel_launch(void* const* d_in, const int* in_sizes, int n_in,
                              void* d_out, int out_size, void* d_ws, size_t ws_size,
                              hipStream_t stream) {
  (void)in_sizes; (void)n_in; (void)out_size; (void)ws_size;
  const float* x     = (const float*)d_in[0];
  const float* W0    = (const float*)d_in[1];
  const float* b0    = (const float*)d_in[2];
  const float* Ws    = (const float*)d_in[3];
  const float* bs    = (const float*)d_in[4];
  const float* bw    = (const float*)d_in[5];
  const float* Wlast = (const float*)d_in[6];
  const float* blast = (const float*)d_in[7];
  const float* noise = (const float*)d_in[8];
  float* y = (float*)d_out;

  // workspace layout (bytes)
  char* ws = (char*)d_ws;
  unsigned short* x_bf = (unsigned short*)(ws + 0);           //  8,388,608
  unsigned short* w0t  = (unsigned short*)(ws + 8388608);     //    262,144
  unsigned short* wst  = (unsigned short*)(ws + 8650752);     // 10,485,760
  float* Smat          = (float*)(ws + 19136512);             //    655,360
  float* outA          = (float*)(ws + 19791872);             // 134,217,728
  float* outB          = (float*)(ws + 154009600);            // 134,217,728

  k_conv_x <<<4096, 256, 0, stream>>>(x, x_bf);
  k_conv_w0<<<512, 256, 0, stream>>>(W0, w0t);
  k_conv_ws<<<8192, 256, 0, stream>>>(Ws, wst);
  k_dots   <<<(NBLK * B_ROWS) / 8, 256, 0, stream>>>(noise, bw, Smat);

  dim3 ggrid(D_DIM / 128, B_ROWS / 128);  // (8, 256)

  // out0 = tanh(x @ W0^T + b0)
  k_gemm<true, false><<<ggrid, 256, 0, stream>>>(
      x_bf, w0t, b0, nullptr, nullptr, nullptr, outA, 128, 128);

  // 5 sequential residual blocks, ping-pong fp32 buffers
  float* src = outA;
  float* dst = outB;
  for (int i = 0; i < NBLK; ++i) {
    k_gemm<false, true><<<ggrid, 256, 0, stream>>>(
        src, wst + (size_t)i * D_DIM * D_DIM, bs + i * D_DIM,
        Smat + i * B_ROWS, bw + i * D_DIM,
        noise + (size_t)i * B_ROWS * D_DIM,
        dst, D_DIM, D_DIM);
    float* t = src; src = dst; dst = t;
  }

  // y = out @ Wlast^T + blast
  k_gemv<<<B_ROWS / 8, 256, 0, stream>>>(src, Wlast, blast, y);
}
